// VectorQuantiser_1391569404581
// MI455X (gfx1250) — matrix operand order repaired and checked by reading, not compiled, not run on hardware
//
#include <hip/hip_runtime.h>
#include <hip/hip_bf16.h>

// ---------- problem constants ----------
#define B_  16
#define T_  2048
#define D_  256
#define K_  1024
#define M_TOTAL (B_ * T_)            // 32768 rows
#define BETA 0.25f

// d_out element offsets (fp32-sized slots, return order: z_q, similarity, ids, loss)
#define ZQ_OFF   0LL
#define SIM_OFF  ((long long)M_TOTAL * D_)             // 8,388,608
#define IDS_OFF  (SIM_OFF + (long long)M_TOTAL * K_)   // 41,943,040
#define LOSS_OFF (IDS_OFF + (long long)M_TOTAL)        // 41,975,808

// d_ws byte offsets
#define WS_CBHI   0
#define WS_CBLO   (WS_CBHI + K_ * D_ * 2)
#define WS_NSQE   (WS_CBLO + K_ * D_ * 2)
#define WS_RINVE  (WS_NSQE + K_ * 4)
#define WS_ROWRES (WS_RINVE + K_ * 4)

typedef __attribute__((ext_vector_type(16))) __bf16 v16bf;
typedef __attribute__((ext_vector_type(8)))  float  v8f;

// ---------------------------------------------------------------------------
// Kernel 1: codebook -> bf16 hi/lo split, code norms + rsqrt norms
// grid = K_, block = 256 (one code row per block). Codebook stays RT in L2.
// ---------------------------------------------------------------------------
__global__ void vq_prep_codebook(const float* __restrict__ cb,
                                 __bf16* __restrict__ cb_hi,
                                 __bf16* __restrict__ cb_lo,
                                 float* __restrict__ norm_sq_e,
                                 float* __restrict__ rinv_e) {
    int k = blockIdx.x;
    int d = threadIdx.x;
    float x = cb[k * D_ + d];
    __bf16 h = (__bf16)x;                   // RNE
    cb_hi[k * D_ + d] = h;
    cb_lo[k * D_ + d] = (__bf16)(x - (float)h);
    __shared__ float red[256];
    red[d] = x * x;
    __syncthreads();
    for (int s = 128; s > 0; s >>= 1) {
        if (d < s) red[d] += red[d + s];
        __syncthreads();
    }
    if (d == 0) {
        norm_sq_e[k] = red[0];
        rinv_e[k] = rsqrtf(red[0]);
    }
}

// ---------------------------------------------------------------------------
// Kernel 2: main VQ kernel. grid = M_TOTAL/16 = 2048 blocks, block = 128 (4 waves).
// Each block owns 16 rows of z_e, sweeps all K=1024 codes with bf16 WMMA
// (split hi/lo -> ~fp32 accuracy, fp32 accumulate, 2 independent WMMA chains).
// ---------------------------------------------------------------------------
__global__ void __launch_bounds__(128)
vq_main(const float* __restrict__ z_e,
        const float* __restrict__ cb,
        const __bf16* __restrict__ cb_hi,
        const __bf16* __restrict__ cb_lo,
        const float* __restrict__ norm_sq_e,
        const float* __restrict__ rinv_e,
        float* __restrict__ zq_out,
        float* __restrict__ sim_out,
        int*   __restrict__ ids_out,
        float* __restrict__ row_res) {
    __shared__ __align__(32) __bf16 a_sw_hi[8 * 32 * 16];  // 8 KB, WMMA-A swizzled
    __shared__ __align__(32) __bf16 a_sw_lo[8 * 32 * 16];  // 8 KB
    __shared__ float af32[16 * 256];                       // 16 KB fp32 copy
    __shared__ float rinv_z[16];
    __shared__ float red_d[4 * 16];
    __shared__ int   red_id[4 * 16];
    __shared__ int   final_id[16];
    __shared__ float partial[128];

    const int m0   = blockIdx.x * 16;
    const int tid  = threadIdx.x;
    const int lane = tid & 31;
    const int wavE = tid >> 5;
    const int half = lane >> 4;
    const int col  = lane & 15;

    // ---- stage z_e tile (fp32, streaming NT read: each row read exactly once) ----
    {
        int m  = tid >> 3;
        int c0 = (tid & 7) * 32;
        const float* src = z_e + (long long)(m0 + m) * D_ + c0;
        #pragma unroll 8
        for (int j = 0; j < 32; ++j)
            af32[m * D_ + c0 + j] = __builtin_nontemporal_load(&src[j]);
    }
    __syncthreads();

    // ---- build bf16 hi/lo tiles in WMMA A-operand order, from LDS (no 2nd global read)
    // chunk c covers K-slice [32c,32c+32); lane L (row m=L&15, half h=L>>4)
    // element j maps to k-offset j + 16h.
    for (int idx = tid; idx < 256; idx += 128) {
        int c = idx >> 5;
        int L = idx & 31;
        int m = L & 15, h = L >> 4;
        #pragma unroll
        for (int j = 0; j < 16; ++j) {
            int d = c * 32 + (j + h * 16);
            float x = af32[m * D_ + d];
            __bf16 hb = (__bf16)x;
            a_sw_hi[idx * 16 + j] = hb;
            a_sw_lo[idx * 16 + j] = (__bf16)(x - (float)hb);
        }
    }
    // ---- per-row 1/||z|| ----
    if (tid < 16) {
        float s = 0.f;
        for (int d = 0; d < D_; ++d) { float x = af32[tid * D_ + d]; s += x * x; }
        rinv_z[tid] = rsqrtf(s);
    }
    __syncthreads();

    float rz[8];
    #pragma unroll
    for (int r = 0; r < 8; ++r) rz[r] = rinv_z[r + half * 8];

    float mind[8];
    int   minid[8];
    #pragma unroll
    for (int r = 0; r < 8; ++r) { mind[r] = 3.4e38f; minid[r] = 0; }

    // ---- sweep 64 N-tiles of 16 codes, 16 per wave ----
    for (int i = 0; i < 16; ++i) {
        int n0   = (i * 4 + wavE) * 16;
        int code = n0 + col;
        const __bf16* bh_base = cb_hi + code * D_ + half * 16;
        const __bf16* bl_base = cb_lo + code * D_ + half * 16;
        v8f acc0 = {};   // hi*hi chain
        v8f acc1 = {};   // hi*lo + lo*hi chain (independent -> WMMA ILP)
        #pragma unroll
        for (int ks = 0; ks < 8; ++ks) {
            v16bf ah = *(const v16bf*)&a_sw_hi[(ks * 32 + lane) * 16];
            v16bf al = *(const v16bf*)&a_sw_lo[(ks * 32 + lane) * 16];
            v16bf bh = *(const v16bf*)(bh_base + ks * 32);
            v16bf bl = *(const v16bf*)(bl_base + ks * 32);
            acc0 = __builtin_amdgcn_wmma_f32_16x16x32_bf16(false, ah, false, bh, (short)0, acc0, false, false);
            acc1 = __builtin_amdgcn_wmma_f32_16x16x32_bf16(false, ah, false, bl, (short)0, acc1, false, false);
            acc1 = __builtin_amdgcn_wmma_f32_16x16x32_bf16(false, al, false, bh, (short)0, acc1, false, false);
        }
        float nE = norm_sq_e[code];
        float rE = rinv_e[code];
        #pragma unroll
        for (int r = 0; r < 8; ++r) {
            int   m   = r + half * 8;
            float dot = acc0[r] + acc1[r];
            // similarity is a 128MB streaming output: NT store keeps codebook in L2
            __builtin_nontemporal_store(dot * rz[r] * rE,
                                        &sim_out[(long long)(m0 + m) * K_ + code]);
            float key = nE - 2.0f * dot;   // argmin-equivalent distance (norm_z const/row)
            if (key < mind[r]) { mind[r] = key; minid[r] = code; }
        }
    }

    // ---- argmin reduce: across 16 lanes of each half (wave32 shuffles) ----
    #pragma unroll
    for (int r = 0; r < 8; ++r) {
        float d = mind[r];
        int   id = minid[r];
        #pragma unroll
        for (int off = 8; off > 0; off >>= 1) {
            float od  = __shfl_xor(d, off, 32);
            int   oid = __shfl_xor(id, off, 32);
            if (od < d || (od == d && oid < id)) { d = od; id = oid; }
        }
        if (col == 0) {
            red_d [wavE * 16 + r + half * 8] = d;
            red_id[wavE * 16 + r + half * 8] = id;
        }
    }
    __syncthreads();

    // ---- argmin reduce across the 4 waves; emit ids ----
    if (tid < 16) {
        int m = tid;
        float bd  = red_d[m];
        int   bid = red_id[m];
        for (int w = 1; w < 4; ++w) {
            float od  = red_d [w * 16 + m];
            int   oid = red_id[w * 16 + m];
            if (od < bd || (od == bd && oid < bid)) { bd = od; bid = oid; }
        }
        final_id[m] = bid;
        ids_out[m0 + m] = bid;
    }
    __syncthreads();

    // ---- gather z_q (NT streaming store), per-row residual L2 norm ----
    {
        int m  = tid >> 3;
        int c0 = (tid & 7) * 32;
        int code = final_id[m];
        const float* crow = cb + (long long)code * D_ + c0;
        float* zqrow = zq_out + (long long)(m0 + m) * D_ + c0;
        float s = 0.f;
        #pragma unroll 8
        for (int j = 0; j < 32; ++j) {
            float q = crow[j];
            __builtin_nontemporal_store(q, &zqrow[j]);
            float dd = af32[m * D_ + c0 + j] - q;
            s += dd * dd;
        }
        partial[tid] = s;
    }
    __syncthreads();
    if (tid < 16) {
        float s = 0.f;
        for (int j = 0; j < 8; ++j) s += partial[tid * 8 + j];
        row_res[m0 + tid] = sqrtf(s);
    }
}

// ---------------------------------------------------------------------------
// Kernel 3: deterministic fixed-order loss reduction.
// loss_vq = (1 + BETA) * mean(||z_e - z_q||_row)   (stop_gradient keeps fwd equal)
// ---------------------------------------------------------------------------
__global__ void vq_loss(const float* __restrict__ row_res, float* __restrict__ loss_out) {
    __shared__ float red[256];
    float s = 0.f;
    for (int i = threadIdx.x; i < M_TOTAL; i += 256) s += row_res[i];
    red[threadIdx.x] = s;
    __syncthreads();
    for (int st = 128; st > 0; st >>= 1) {
        if (threadIdx.x < st) red[threadIdx.x] += red[threadIdx.x + st];
        __syncthreads();
    }
    if (threadIdx.x == 0) loss_out[0] = (1.0f + BETA) * red[0] / (float)M_TOTAL;
}

// ---------------------------------------------------------------------------
extern "C" void kernel_launch(void* const* d_in, const int* in_sizes, int n_in,
                              void* d_out, int out_size, void* d_ws, size_t ws_size,
                              hipStream_t stream) {
    const float* z_e = (const float*)d_in[0];
    const float* cb  = (const float*)d_in[1];

    char* ws = (char*)d_ws;
    __bf16* cb_hi      = (__bf16*)(ws + WS_CBHI);
    __bf16* cb_lo      = (__bf16*)(ws + WS_CBLO);
    float*  norm_sq_e  = (float*)(ws + WS_NSQE);
    float*  rinv_e     = (float*)(ws + WS_RINVE);
    float*  row_res    = (float*)(ws + WS_ROWRES);

    float* outf    = (float*)d_out;
    float* zq_out  = outf + ZQ_OFF;
    float* sim_out = outf + SIM_OFF;
    int*   ids_out = (int*)(outf + IDS_OFF);
    float* loss_out = outf + LOSS_OFF;

    vq_prep_codebook<<<K_, 256, 0, stream>>>(cb, cb_hi, cb_lo, norm_sq_e, rinv_e);
    vq_main<<<M_TOTAL / 16, 128, 0, stream>>>(z_e, cb, cb_hi, cb_lo, norm_sq_e, rinv_e,
                                              zq_out, sim_out, ids_out, row_res);
    vq_loss<<<1, 256, 0, stream>>>(row_res, loss_out);
}
